// Guide_35570919146244
// MI455X (gfx1250) — matrix operand order repaired and checked by reading, not compiled, not run on hardware
//
#include <hip/hip_runtime.h>

#define N_  2
#define C_  64
#define WP_ 64
#define H_  240
#define W_  320
#define NB_ 6
#define HW_ (H_*W_)
#define EPS_ 1e-5f

typedef __attribute__((ext_vector_type(16))) _Float16     v16h;
typedef __attribute__((ext_vector_type(8)))  float        v8f;
typedef __attribute__((ext_vector_type(4)))  unsigned int u32x4;
typedef __attribute__((ext_vector_type(8)))  int          i32x8;
typedef __attribute__((ext_vector_type(4)))  int          i32x4;

// ---------------------------------------------------------------------------
// One-shot weight repack into WMMA-friendly f16 layouts (k-contiguous):
//   w1h[tap][o][c]  (9*64*128)   w3h[tap][o][c] (9*64*64)   coefh[o][k] (64*384)
// ---------------------------------------------------------------------------
__global__ __launch_bounds__(256)
void repack_kernel(const float* __restrict__ w1, const float* __restrict__ w3,
                   const float* __restrict__ coef,
                   _Float16* __restrict__ w1h, _Float16* __restrict__ w3h,
                   _Float16* __restrict__ coefh)
{
    const int i = blockIdx.x * 256 + threadIdx.x;
    if (i < 9 * 64 * 128) {
        const int c = i % 128, o = (i / 128) % 64, tap = i / (128 * 64);
        w1h[i] = (_Float16)w1[((size_t)o * 128 + c) * 9 + tap];
    }
    if (i < 9 * 64 * 64) {
        const int c = i % 64, o = (i / 64) % 64, tap = i / (64 * 64);
        w3h[i] = (_Float16)w3[((size_t)o * 64 + c) * 9 + tap];
    }
    if (i < 64 * 384) coefh[i] = (_Float16)coef[i];
}

// ---------------------------------------------------------------------------
// Kernel 1: conv1 3x3 over concat(feat,weight) 128->64 + bias + ReLU  (WMMA)
// Halo tile staged in LDS as [3 rows][18 cols][128 ch] f16; A = 2x ds_load_b128
// ---------------------------------------------------------------------------
__global__ __launch_bounds__(128)
void conv1_kernel(const float* __restrict__ feat, const float* __restrict__ wgt,
                  const _Float16* __restrict__ w1h, const float* __restrict__ b1,
                  _Float16* __restrict__ t_out)
{
    __shared__ _Float16 ash[3 * 18 * 128];        // 13824 B

    const int tilesPerRow = W_ / 16;
    int t = blockIdx.x;
    const int tw = t % tilesPerRow; t /= tilesPerRow;
    const int h  = t % H_;          t /= H_;
    const int n  = t;
    const int w0 = tw * 16;
    const int tid = threadIdx.x;

    const float* in0 = feat + (size_t)n * C_  * HW_;
    const float* in1 = wgt  + (size_t)n * WP_ * HW_;

    // stage halo: clamp address in-bounds, select 0 out-of-bounds (branchless)
    for (int i = 0; i < 54; ++i) {
        const int e    = tid + i * 128;
        const int col  = e % 18;
        const int c    = (e / 18) % 128;
        const int rowi = e / (18 * 128);
        const int y  = h + rowi - 1;
        const int x  = w0 + col - 1;
        const int yc = min(max(y, 0), H_ - 1);
        const int xc = min(max(x, 0), W_ - 1);
        const float* src = (c < C_) ? (in0 + (size_t)c * HW_)
                                    : (in1 + (size_t)(c - C_) * HW_);
        const float v  = src[yc * W_ + xc];
        const bool inb = (y >= 0) && (y < H_) && (x >= 0) && (x < W_);
        ash[(rowi * 18 + col) * 128 + c] = (_Float16)(inb ? v : 0.0f);
    }
    __syncthreads();

    const int lane = tid & 31;
    const int wave = tid >> 5;
    const int row  = lane & 15;       // pixel within tile (A row)
    const int hi   = lane >> 4;
    const int kbA  = hi ? 16 : 0;
    const int kbB  = hi ? 16 : 0;
    const int o    = wave * 16 + row; // output channel (B column)

    v8f acc = {};
    for (int tap = 0; tap < 9; ++tap) {
        const int dy = tap / 3, dx = tap % 3;
        const _Float16* arow = &ash[(dy * 18 + row + dx) * 128];
        const _Float16* brow = w1h + ((size_t)tap * 64 + o) * 128;
        if (tap < 8) __builtin_prefetch(brow + 64 * 128, 0, 0);
        for (int kc = 0; kc < 4; ++kc) {
            union { uint4 q[2]; v16h v; } av, bv;
            av.q[0] = *reinterpret_cast<const uint4*>(arow + kc * 32 + kbA);
            av.q[1] = *reinterpret_cast<const uint4*>(arow + kc * 32 + 8 + kbA);
            bv.q[0] = *reinterpret_cast<const uint4*>(brow + kc * 32 + kbB);
            bv.q[1] = *reinterpret_cast<const uint4*>(brow + kc * 32 + kbB + 8);
            acc = __builtin_amdgcn_wmma_f32_16x16x32_f16(false, av.v, false, bv.v,
                                                         (short)0, acc, false, false);
        }
    }
    const float bias = b1[o];
    alignas(16) _Float16 ov[8];
#pragma unroll
    for (int v = 0; v < 8; ++v) {
        const float r = acc[v] + bias;
        ov[v] = (_Float16)(r > 0.0f ? r : 0.0f);
    }
    _Float16* dst = t_out + ((size_t)(n * C_ + o) * H_ + h) * W_ + w0 + hi * 8;
    *reinterpret_cast<uint4*>(dst) = *reinterpret_cast<const uint4*>(ov);
}

// ---------------------------------------------------------------------------
// Kernel 2: conv2 1x1 64->54 (tiny). t-tile (64 ch x 64 pix) fetched by the
// Tensor Data Mover (TDM) straight into LDS; w2 loaded while DMA is in flight.
// ---------------------------------------------------------------------------
__global__ __launch_bounds__(256)
void conv2_kernel(const _Float16* __restrict__ t_in,
                  const float* __restrict__ w2, const float* __restrict__ b2,
                  _Float16* __restrict__ bases)
{
    __shared__ _Float16 tsh[C_ * 64];     // [c][p], x-fastest == TDM tile order
    __shared__ float    wsh[54 * C_];

    const int pixBlocks = HW_ / 64;       // 1200
    const int n    = blockIdx.x / pixBlocks;
    const int pix0 = (blockIdx.x % pixBlocks) * 64;
    const int tid  = threadIdx.x;

    // --- TDM descriptor: 2D tile, dim0 = 64 pixels (contig), dim1 = 64 ch ---
    {
        const unsigned long long ga =
            (unsigned long long)(const void*)(t_in + (size_t)n * C_ * HW_ + pix0);
        const unsigned int la =
            (unsigned int)(unsigned long long)(const void*)&tsh[0];
        u32x4 g0 = { 1u,                                   // count=1, user desc
                     la,                                   // lds_addr
                     (unsigned int)(ga & 0xFFFFFFFFu),     // global_addr[31:0]
                     (unsigned int)(((ga >> 32) & 0x01FFFFFFu) | (2u << 30)) };
        i32x8 g1 = { (int)(1u << 16),      // data_size=1 -> 2 bytes/elem
                     (int)(64u << 16),     // tensor_dim0[15:0]=64
                     (int)(64u << 16),     // tensor_dim0 hi=0 | tensor_dim1 lo=64
                     (int)(64u << 16),     // tensor_dim1 hi=0 | tile_dim0=64
                     (int)(64u),           // tile_dim1=64 | tile_dim2=0
                     (int)(HW_),           // tensor_dim0_stride lo = 76800
                     0, 0 };
        i32x4 gz4 = { 0, 0, 0, 0 };
        i32x8 gz8 = { 0, 0, 0, 0, 0, 0, 0, 0 };
        __builtin_amdgcn_tensor_load_to_lds(g0, g1, gz4, gz4, gz8, 0);
    }

    // overlap: cooperative load of w2 while the tensor DMA runs
    for (int i = 0; i < 14; ++i) {
        const int e = tid + i * 256;
        if (e < 54 * C_) wsh[e] = w2[e];
    }
    __builtin_amdgcn_s_wait_tensorcnt(0);
    __syncthreads();

    for (int i = 0; i < 14; ++i) {        // 54 oc x 64 pix outputs
        const int e = tid + i * 256;
        if (e >= 54 * 64) break;
        const int oc = e >> 6, p = e & 63;
        float s = b2[oc];
#pragma unroll 8
        for (int c = 0; c < C_; ++c)
            s += (float)tsh[c * 64 + p] * wsh[oc * C_ + c];
        bases[((size_t)(n * 54 + oc)) * HW_ + pix0 + p] = (_Float16)s;
    }
}

// ---------------------------------------------------------------------------
// Kernel 3 (fused): feat halo + bases staged in LDS; per-pixel einsum to
// bases_out[16][384] f16 in LDS; coef GEMM via WMMA (K=384) + BN + ReLU
// ---------------------------------------------------------------------------
__global__ __launch_bounds__(128)
void fused_coef_kernel(const float* __restrict__ feat,
                       const _Float16* __restrict__ bases,
                       const _Float16* __restrict__ coefh,
                       const float* __restrict__ cbias,
                       const float* __restrict__ brg, const float* __restrict__ brb,
                       const float* __restrict__ brm, const float* __restrict__ brv,
                       _Float16* __restrict__ mid)
{
    __shared__ float    psh[3 * 18 * C_];    // feat halo, 13824 B
    __shared__ float    bsh[16][54];         // bases per pixel
    __shared__ _Float16 bosh[16][NB_ * C_];  // bases_out, 768 B rows

    const int tilesPerRow = W_ / 16;
    int t = blockIdx.x;
    const int tw = t % tilesPerRow; t /= tilesPerRow;
    const int h  = t % H_;          t /= H_;
    const int n  = t;
    const int w0 = tw * 16;
    const int tid = threadIdx.x;

    const float* f0 = feat + (size_t)n * C_ * HW_;
    for (int i = 0; i < 27; ++i) {           // 3*18*64 halo elements
        const int e    = tid + i * 128;
        const int col  = e % 18;
        const int c    = (e / 18) % C_;
        const int rowi = e / (18 * C_);
        const int y  = h + rowi - 1;
        const int x  = w0 + col - 1;
        const int yc = min(max(y, 0), H_ - 1);
        const int xc = min(max(x, 0), W_ - 1);
        const float v  = f0[(size_t)c * HW_ + yc * W_ + xc];
        const bool inb = (y >= 0) && (y < H_) && (x >= 0) && (x < W_);
        psh[(rowi * 18 + col) * C_ + c] = inb ? v : 0.0f;
    }
    for (int i = 0; i < 7; ++i) {            // 16 pix x 54 bases
        const int e = tid + i * 128;
        if (e < 16 * 54) {
            const int p = e / 54, oc = e % 54;
            bsh[p][oc] = (float)bases[((size_t)(n * 54 + oc) * H_ + h) * W_ + w0 + p];
        }
    }
    __syncthreads();

    // einsum: bases_out[p][c*6+m] = sum_l psh[c][l@p] * bsh[p][m*9+l]
    {
        const int c  = tid >> 1;
        const int m0 = (tid & 1) * 3;
        for (int p = 0; p < 16; ++p) {
            float pv[9];
#pragma unroll
            for (int l = 0; l < 9; ++l)
                pv[l] = psh[((l / 3) * 18 + p + (l % 3)) * C_ + c];
#pragma unroll
            for (int mm = 0; mm < 3; ++mm) {
                const int m = m0 + mm;
                float s = 0.0f;
#pragma unroll
                for (int l = 0; l < 9; ++l) s += pv[l] * bsh[p][m * 9 + l];
                bosh[p][c * NB_ + m] = (_Float16)s;
            }
        }
    }
    __syncthreads();

    const int lane = tid & 31;
    const int wave = tid >> 5;
    const int row  = lane & 15;
    const int hi   = lane >> 4;
    const int kbA  = hi ? 16 : 0;
    const int kbB  = hi ? 16 : 0;
    const int o    = wave * 16 + row;

    v8f acc = {};
    for (int kc = 0; kc < 12; ++kc) {
        union { uint4 q[2]; v16h v; } av, bv;
        av.q[0] = *reinterpret_cast<const uint4*>(&bosh[row][kc * 32 + kbA]);
        av.q[1] = *reinterpret_cast<const uint4*>(&bosh[row][kc * 32 + 8 + kbA]);
        const _Float16* bp = coefh + (size_t)o * 384 + kc * 32 + kbB;
        bv.q[0] = *reinterpret_cast<const uint4*>(bp);
        bv.q[1] = *reinterpret_cast<const uint4*>(bp + 8);
        acc = __builtin_amdgcn_wmma_f32_16x16x32_f16(false, av.v, false, bv.v,
                                                     (short)0, acc, false, false);
    }

    const float bias = cbias[o];
    const float inv  = brg[o] * rsqrtf(brv[o] + EPS_);
    const float bet  = brb[o] - brm[o] * inv;
    alignas(16) _Float16 ov[8];
#pragma unroll
    for (int v = 0; v < 8; ++v) {
        const float r = (acc[v] + bias) * inv + bet;
        ov[v] = (_Float16)(r > 0.0f ? r : 0.0f);
    }
    _Float16* dst = mid + ((size_t)(n * C_ + o) * H_ + h) * W_ + w0 + hi * 8;
    *reinterpret_cast<uint4*>(dst) = *reinterpret_cast<const uint4*>(ov);
}

// ---------------------------------------------------------------------------
// Kernel 4: conv3 3x3 64->64 + BN + ReLU -> d_out f32  (WMMA, LDS-staged halo)
// ---------------------------------------------------------------------------
__global__ __launch_bounds__(128)
void conv3_kernel(const _Float16* __restrict__ mid, const _Float16* __restrict__ w3h,
                  const float* __restrict__ g3, const float* __restrict__ b3,
                  const float* __restrict__ m3, const float* __restrict__ v3,
                  float* __restrict__ out)
{
    __shared__ _Float16 ash[3 * 18 * C_];     // 6912 B

    const int tilesPerRow = W_ / 16;
    int t = blockIdx.x;
    const int tw = t % tilesPerRow; t /= tilesPerRow;
    const int h  = t % H_;          t /= H_;
    const int n  = t;
    const int w0 = tw * 16;
    const int tid = threadIdx.x;

    const _Float16* in = mid + (size_t)n * C_ * HW_;
    for (int i = 0; i < 27; ++i) {
        const int e    = tid + i * 128;
        const int col  = e % 18;
        const int c    = (e / 18) % C_;
        const int rowi = e / (18 * C_);
        const int y  = h + rowi - 1;
        const int x  = w0 + col - 1;
        const int yc = min(max(y, 0), H_ - 1);
        const int xc = min(max(x, 0), W_ - 1);
        const _Float16 v = in[(size_t)c * HW_ + yc * W_ + xc];
        const bool inb = (y >= 0) && (y < H_) && (x >= 0) && (x < W_);
        ash[(rowi * 18 + col) * C_ + c] = inb ? v : (_Float16)0.0f;
    }
    __syncthreads();

    const int lane = tid & 31;
    const int wave = tid >> 5;
    const int row  = lane & 15;
    const int hi   = lane >> 4;
    const int kbA  = hi ? 16 : 0;
    const int kbB  = hi ? 16 : 0;
    const int o    = wave * 16 + row;

    v8f acc = {};
    for (int tap = 0; tap < 9; ++tap) {
        const int dy = tap / 3, dx = tap % 3;
        const _Float16* arow = &ash[(dy * 18 + row + dx) * C_];
        const _Float16* brow = w3h + ((size_t)tap * 64 + o) * C_;
        if (tap < 8) __builtin_prefetch(brow + 64 * C_, 0, 0);
        for (int kc = 0; kc < 2; ++kc) {
            union { uint4 q[2]; v16h v; } av, bv;
            av.q[0] = *reinterpret_cast<const uint4*>(arow + kc * 32 + kbA);
            av.q[1] = *reinterpret_cast<const uint4*>(arow + kc * 32 + 8 + kbA);
            bv.q[0] = *reinterpret_cast<const uint4*>(brow + kc * 32 + kbB);
            bv.q[1] = *reinterpret_cast<const uint4*>(brow + kc * 32 + kbB + 8);
            acc = __builtin_amdgcn_wmma_f32_16x16x32_f16(false, av.v, false, bv.v,
                                                         (short)0, acc, false, false);
        }
    }
    const float inv = g3[o] * rsqrtf(v3[o] + EPS_);
    const float bet = b3[o] - m3[o] * inv;
    alignas(16) float ov[8];
#pragma unroll
    for (int v = 0; v < 8; ++v) {
        const float r = acc[v] * inv + bet;
        ov[v] = (r > 0.0f ? r : 0.0f);
    }
    float* dst = out + ((size_t)(n * C_ + o) * H_ + h) * W_ + w0 + hi * 8;
    reinterpret_cast<float4*>(dst)[0] = reinterpret_cast<const float4*>(ov)[0];
    reinterpret_cast<float4*>(dst)[1] = reinterpret_cast<const float4*>(ov)[1];
}

// ---------------------------------------------------------------------------
extern "C" void kernel_launch(void* const* d_in, const int* in_sizes, int n_in,
                              void* d_out, int out_size, void* d_ws, size_t ws_size,
                              hipStream_t stream)
{
    const float* feat = (const float*)d_in[0];
    const float* wgt  = (const float*)d_in[1];
    const float* w1   = (const float*)d_in[2];
    const float* b1   = (const float*)d_in[3];
    const float* w2   = (const float*)d_in[4];
    const float* b2   = (const float*)d_in[5];
    const float* coef = (const float*)d_in[6];
    const float* cb   = (const float*)d_in[7];
    const float* brg  = (const float*)d_in[8];
    const float* brb  = (const float*)d_in[9];
    const float* brm  = (const float*)d_in[10];
    const float* brv  = (const float*)d_in[11];
    const float* w3   = (const float*)d_in[12];
    const float* g3   = (const float*)d_in[13];
    const float* b3   = (const float*)d_in[14];
    const float* m3   = (const float*)d_in[15];
    const float* v3   = (const float*)d_in[16];
    float* out = (float*)d_out;

    _Float16* t16   = (_Float16*)d_ws;                    // [N,64,H,W]
    _Float16* bases = t16   + (size_t)N_ * C_ * HW_;      // [N,54,H,W]
    _Float16* mid   = bases + (size_t)N_ * 54 * HW_;      // [N,64,H,W]
    _Float16* w1h   = mid   + (size_t)N_ * C_ * HW_;      // [9,64,128]
    _Float16* w3h   = w1h   + (size_t)9 * 64 * 128;       // [9,64,64]
    _Float16* coefh = w3h   + (size_t)9 * 64 * 64;        // [64,384]

    const int tiles = N_ * H_ * (W_ / 16);                // 9600

    repack_kernel    <<<(9 * 64 * 128 + 255) / 256, 256, 0, stream>>>(w1, w3, coef,
                                                                      w1h, w3h, coefh);
    conv1_kernel     <<<tiles,           128, 0, stream>>>(feat, wgt, w1h, b1, t16);
    conv2_kernel     <<<N_ * (HW_ / 64), 256, 0, stream>>>(t16, w2, b2, bases);
    fused_coef_kernel<<<tiles,           128, 0, stream>>>(feat, bases, coefh, cb,
                                                           brg, brb, brm, brv, mid);
    conv3_kernel     <<<tiles,           128, 0, stream>>>(mid, w3h, g3, b3, m3, v3, out);
}
